// GCN_67207648248072
// MI455X (gfx1250) — matrix operand order repaired and checked by reading, not compiled, not run on hardware
//
#include <hip/hip_runtime.h>

typedef __attribute__((ext_vector_type(16))) __bf16    v16bf;
typedef __attribute__((ext_vector_type(8)))  float     v8f;
typedef __attribute__((ext_vector_type(8)))  unsigned  v8u;

// ---------- f32 -> bf16 split helpers (round-to-nearest-even) ----------
__device__ __forceinline__ unsigned f2bf(float f) {
  unsigned u = __builtin_bit_cast(unsigned, f);
  return (u + 0x7FFFu + ((u >> 16) & 1u)) >> 16;
}
__device__ __forceinline__ float bf2f(unsigned h) {
  return __builtin_bit_cast(float, h << 16);
}

// ---------------------------------------------------------------------
// Pack weight matrix W[K x N] (f32, row-major) into WMMA B-fragment image.
// Fragment for (kb, t, lane) = 16 dwords: [0..7] = hi bf16 pairs, [8..15] = lo.
// B layout (32x16, 16-bit): lane n<16 -> col n, K={0..7,16..23} ; lane n+16 -> col n, K={8..15,24..31}.
// ---------------------------------------------------------------------
__global__ void pack_w_kernel(const float* __restrict__ W, unsigned* __restrict__ Wp,
                              int K, int N) {
  int nt    = N >> 4;
  int total = (K >> 5) * nt * 32;
  int tid   = blockIdx.x * blockDim.x + threadIdx.x;
  if (tid >= total) return;
  int lane = tid & 31;
  int t    = (tid >> 5) % nt;
  int kb   = tid / (32 * nt);
  int col  = t * 16 + (lane & 15);
  int k0   = kb * 32 + ((lane < 16) ? 0 : 8);
  unsigned* out = Wp + (size_t)tid * 16;
#pragma unroll
  for (int j = 0; j < 8; ++j) {
    float x = W[(size_t)(k0 + 2 * j + (j & 4) * 2)     * N + col];
    float y = W[(size_t)(k0 + 2 * j + (j & 4) * 2 + 1) * N + col];
    unsigned hx = f2bf(x), hy = f2bf(y);
    out[j]     = hx | (hy << 16);
    out[8 + j] = f2bf(x - bf2f(hx)) | (f2bf(y - bf2f(hy)) << 16);
  }
}

// ---------------------------------------------------------------------
// C[M x ldc](f32) = A[M x K](f32) * W (pre-packed bf16 hi/lo fragments),
// computed as 3x bf16 WMMA (hi*hi + hi*lo + lo*hi) per 16x16x32 step.
// Block: 256 threads = 8 waves; 128x128 C tile; wave w -> rows w*16..w*16+15.
// ---------------------------------------------------------------------
__global__ __launch_bounds__(256)
void gemm_bf16x2_kernel(const float* __restrict__ A, const unsigned* __restrict__ Wp,
                        float* __restrict__ C, int M, int K, int totNT, int ldc) {
  const int wave   = threadIdx.x >> 5;
  const int lane   = threadIdx.x & 31;
  const int laneLo = lane & 15;
  const int nbase  = blockIdx.y * 8;
  const int mBase  = blockIdx.x * 128 + wave * 16;

  // A-fragment row for this lane (16-bit A 16x32 layout), clamped for tail block
  int aRow = mBase + laneLo;
  if (aRow >= M) aRow = M - 1;
  const float* aPtr = A + (size_t)aRow * K + ((lane < 16) ? 0 : 8);

  v8f acc[8];
#pragma unroll
  for (int t = 0; t < 8; ++t) acc[t] = (v8f)0.0f;

  const int nkb = K >> 5;
  for (int kb = 0; kb < nkb; ++kb) {
    // lane<16 needs K{0..7,16..23}; lane>=16 needs K{8..15,24..31} (base shift +8)
    const float4* ap = (const float4*)(aPtr + (size_t)kb * 32);
    float4 f0 = ap[0];
    float4 f1 = ap[1];
    float4 f2 = ap[4];
    float4 f3 = ap[5];
    if (kb + 1 < nkb) __builtin_prefetch(aPtr + (size_t)(kb + 1) * 32, 0, 0);

    float av[16] = {f0.x, f0.y, f0.z, f0.w, f1.x, f1.y, f1.z, f1.w,
                    f2.x, f2.y, f2.z, f2.w, f3.x, f3.y, f3.z, f3.w};
    v8u hu, lu;
#pragma unroll
    for (int j = 0; j < 8; ++j) {
      unsigned h0 = f2bf(av[2 * j]);
      unsigned h1 = f2bf(av[2 * j + 1]);
      unsigned l0 = f2bf(av[2 * j]     - bf2f(h0));
      unsigned l1 = f2bf(av[2 * j + 1] - bf2f(h1));
      hu[j] = h0 | (h1 << 16);
      lu[j] = l0 | (l1 << 16);
    }
    v16bf a_hi = __builtin_bit_cast(v16bf, hu);
    v16bf a_lo = __builtin_bit_cast(v16bf, lu);

    const size_t wIdxBase = ((size_t)kb * totNT + nbase) * 32 + lane;
#pragma unroll
    for (int t = 0; t < 8; ++t) {
      const uint4* bp = (const uint4*)(Wp + (wIdxBase + (size_t)t * 32) * 16);
      uint4 bh0 = bp[0], bh1 = bp[1], bl0 = bp[2], bl1 = bp[3];
      v8u bhu = {bh0.x, bh0.y, bh0.z, bh0.w, bh1.x, bh1.y, bh1.z, bh1.w};
      v8u blu = {bl0.x, bl0.y, bl0.z, bl0.w, bl1.x, bl1.y, bl1.z, bl1.w};
      v16bf b_hi = __builtin_bit_cast(v16bf, bhu);
      v16bf b_lo = __builtin_bit_cast(v16bf, blu);
      acc[t] = __builtin_amdgcn_wmma_f32_16x16x32_bf16(false, a_hi, false, b_hi,
                                                       (short)0, acc[t], false, false);
      acc[t] = __builtin_amdgcn_wmma_f32_16x16x32_bf16(false, a_hi, false, b_lo,
                                                       (short)0, acc[t], false, false);
      acc[t] = __builtin_amdgcn_wmma_f32_16x16x32_bf16(false, a_lo, false, b_hi,
                                                       (short)0, acc[t], false, false);
    }
  }

  // C/D layout: VGPR j -> row mBase + j (lanes 0-15) / mBase + 8 + j (lanes 16-31)
  const int rBase = mBase + ((lane < 16) ? 0 : 8);
  const int col0  = blockIdx.y * 128 + laneLo;
#pragma unroll
  for (int j = 0; j < 8; ++j) {
    int r = rBase + j;
    if (r < M) {
      float* crow = C + (size_t)r * ldc + col0;
#pragma unroll
      for (int t = 0; t < 8; ++t) crow[t * 16] = acc[t][j];
    }
  }
}

// ---------------------------------------------------------------------
// Graph aggregation helpers
// ---------------------------------------------------------------------
__global__ void zero_f32_kernel(float* __restrict__ p, int n) {
  int i = blockIdx.x * blockDim.x + threadIdx.x;
  if (i < n) p[i] = 0.0f;
}

__global__ void deg_kernel(const long long* __restrict__ ei, int E,
                           float* __restrict__ deg) {
  int e = blockIdx.x * blockDim.x + threadIdx.x;
  if (e < E) atomicAdd(&deg[ei[E + e]], 1.0f);  // dst row of edge_index
}

__global__ void dinv_kernel(const float* __restrict__ deg, float* __restrict__ dinv, int n) {
  int i = blockIdx.x * blockDim.x + threadIdx.x;
  if (i < n) dinv[i] = rsqrtf(deg[i] + 1.0f);  // +1 for self loop; always > 0
}

// agg[i][f] = dinv[i]^2 * h[i][f] + bias[f]   (self-loop term + bias)
__global__ void init_agg_kernel(const float* __restrict__ h, const float* __restrict__ dinv,
                                const float* __restrict__ bias, float* __restrict__ agg,
                                int M, int F) {
  int i = blockIdx.x * blockDim.x + threadIdx.x;
  if (i >= M * F) return;
  int r = i / F, f = i - r * F;
  float d = dinv[r];
  agg[i] = d * d * h[i] + bias[f];
}

// One wave per edge; each lane handles 4 features per iteration.
__global__ void scatter_kernel(const long long* __restrict__ ei, int E,
                               const float* __restrict__ dinv,
                               const float* __restrict__ h, float* __restrict__ agg,
                               int F) {
  int gw   = (blockIdx.x * blockDim.x + threadIdx.x) >> 5;
  int lane = threadIdx.x & 31;
  if (gw >= E) return;
  long long s = ei[gw];
  long long d = ei[E + gw];
  float w = dinv[s] * dinv[d];
  const float4* src = (const float4*)(h + (size_t)s * F);
  float* dst = agg + (size_t)d * F;
  int nIter = F >> 7;  // 128 features per wave-iteration
  for (int it = 0; it < nIter; ++it) {
    int c = it * 32 + lane;
    float4 v = src[c];
    atomicAdd(dst + 4 * c + 0, w * v.x);
    atomicAdd(dst + 4 * c + 1, w * v.y);
    atomicAdd(dst + 4 * c + 2, w * v.z);
    atomicAdd(dst + 4 * c + 3, w * v.w);
  }
}

__global__ void relu_kernel(float* __restrict__ p, int n) {
  int i = blockIdx.x * blockDim.x + threadIdx.x;
  if (i < n) p[i] = fmaxf(p[i], 0.0f);
}

// ---------------------------------------------------------------------
extern "C" void kernel_launch(void* const* d_in, const int* in_sizes, int n_in,
                              void* d_out, int out_size, void* d_ws, size_t ws_size,
                              hipStream_t stream) {
  (void)n_in; (void)out_size; (void)ws_size;
  const float*     x   = (const float*)d_in[0];
  const long long* ei  = (const long long*)d_in[1];
  const float*     W1  = (const float*)d_in[2];
  const float*     b1  = (const float*)d_in[3];
  const float*     W2  = (const float*)d_in[4];
  const float*     b2  = (const float*)d_in[5];
  float*           out = (float*)d_out;

  const int HID    = in_sizes[3];              // 128
  const int IN_DIM = in_sizes[2] / HID;        // 27136
  const int OUT    = in_sizes[5];              // 768
  const int NN     = in_sizes[0] / IN_DIM;     // 10000
  const int E      = in_sizes[1] / 2;          // 320000

  // Workspace carve (256B aligned)
  char* p = (char*)d_ws;
  auto alignUp = [](size_t v) { return (v + 255) & ~(size_t)255; };
  float* deg  = (float*)p; p += alignUp((size_t)NN * 4);
  float* dinv = (float*)p; p += alignUp((size_t)NN * 4);
  float* h0   = (float*)p; p += alignUp((size_t)NN * HID * 4);
  float* agg1 = (float*)p; p += alignUp((size_t)NN * HID * 4);
  float* h1   = (float*)p; p += alignUp((size_t)NN * OUT * 4);
  unsigned* W1p = (unsigned*)p; p += alignUp((size_t)(IN_DIM / 32) * (HID / 16) * 32 * 16 * 4);
  unsigned* W2p = (unsigned*)p; p += alignUp((size_t)(HID / 32) * (OUT / 16) * 32 * 16 * 4);

  const int T = 256;
  // Degree / normalization
  zero_f32_kernel<<<(NN + T - 1) / T, T, 0, stream>>>(deg, NN);
  deg_kernel<<<(E + T - 1) / T, T, 0, stream>>>(ei, E, deg);
  dinv_kernel<<<(NN + T - 1) / T, T, 0, stream>>>(deg, dinv, NN);

  // Pre-pack weights into bf16 hi/lo B-fragment images
  int w1frag = (IN_DIM / 32) * (HID / 16) * 32;
  int w2frag = (HID / 32) * (OUT / 16) * 32;
  pack_w_kernel<<<(w1frag + T - 1) / T, T, 0, stream>>>(W1, W1p, IN_DIM, HID);
  pack_w_kernel<<<(w2frag + T - 1) / T, T, 0, stream>>>(W2, W2p, HID, OUT);

  // Layer 1: h0 = x @ W1 ; agg1 = scatter(norm * h0) + b1 ; relu in place
  gemm_bf16x2_kernel<<<dim3((NN + 127) / 128, HID / 128), T, 0, stream>>>(
      x, W1p, h0, NN, IN_DIM, HID / 16, HID);
  init_agg_kernel<<<(NN * HID + T - 1) / T, T, 0, stream>>>(h0, dinv, b1, agg1, NN, HID);
  scatter_kernel<<<(E * 32 + T - 1) / T, T, 0, stream>>>(ei, E, dinv, h0, agg1, HID);
  relu_kernel<<<(NN * HID + T - 1) / T, T, 0, stream>>>(agg1, NN * HID);

  // Layer 2: h1 = relu_agg @ W2 ; out = scatter(norm * h1) + b2
  gemm_bf16x2_kernel<<<dim3((NN + 127) / 128, OUT / 128), T, 0, stream>>>(
      agg1, W2p, h1, NN, HID, OUT / 16, OUT);
  init_agg_kernel<<<(NN * OUT + T - 1) / T, T, 0, stream>>>(h1, dinv, b2, out, NN, OUT);
  scatter_kernel<<<(E * 32 + T - 1) / T, T, 0, stream>>>(ei, E, dinv, h1, out, OUT);
}
